// mLSTMCell_59657095741677
// MI455X (gfx1250) — matrix operand order repaired and checked by reading, not compiled, not run on hardware
//
#include <hip/hip_runtime.h>
#include <math.h>
#include <stdint.h>

#define B_  2
#define S_  2048
#define H_  1024
#define NH_ 4
#define DH_ 256

typedef __attribute__((ext_vector_type(16))) _Float16 v16h;
typedef __attribute__((ext_vector_type(8)))  _Float16 v8h;
typedef __attribute__((ext_vector_type(4)))  _Float16 v4h;
typedef __attribute__((ext_vector_type(8)))  float    v8f;
typedef uint32_t u32;
typedef u32 u32x4 __attribute__((ext_vector_type(4)));
typedef u32 u32x8 __attribute__((ext_vector_type(8)));

// ---------------------------------------------------------------------------
// Kernel 1: gate pre-activations.  ig/fg[b,s,h] = concat(q,k,v)[b,s,:] @ W[:,h] + bias[h]
// ---------------------------------------------------------------------------
__global__ void gates_kernel(const float* __restrict__ q, const float* __restrict__ k,
                             const float* __restrict__ v,
                             const float* __restrict__ igk, const float* __restrict__ igb,
                             const float* __restrict__ fgk, const float* __restrict__ fgb,
                             float* __restrict__ igp, float* __restrict__ fgp)
{
  const int bs  = blockIdx.x;            // b*S + s
  const int b   = bs / S_, s = bs % S_;
  const int tid = threadIdx.x;           // 128
  const size_t rowbase = (size_t)bs * H_;

  float aI[NH_] = {0.f,0.f,0.f,0.f};
  float aF[NH_] = {0.f,0.f,0.f,0.f};
  #pragma unroll
  for (int i = 0; i < 24; ++i) {         // 24*128 = 3072 features
    const int f = i*128 + tid;
    float val;
    if (f < H_)          val = q[rowbase + f];
    else if (f < 2*H_)   val = k[rowbase + f - H_];
    else                 val = v[rowbase + f - 2*H_];
    const float4 ik = ((const float4*)igk)[f];   // (3H, NH=4) row-major
    const float4 fk = ((const float4*)fgk)[f];
    aI[0] += val*ik.x; aI[1] += val*ik.y; aI[2] += val*ik.z; aI[3] += val*ik.w;
    aF[0] += val*fk.x; aF[1] += val*fk.y; aF[2] += val*fk.z; aF[3] += val*fk.w;
  }
  __shared__ float red[8][128];
  #pragma unroll
  for (int j = 0; j < 4; ++j) { red[j][tid] = aI[j]; red[4+j][tid] = aF[j]; }
  __syncthreads();
  for (int st = 64; st > 0; st >>= 1) {
    if (tid < st) {
      #pragma unroll
      for (int j = 0; j < 8; ++j) red[j][tid] += red[j][tid+st];
    }
    __syncthreads();
  }
  if (tid < NH_) {
    igp[((size_t)(b*NH_ + tid))*S_ + s] = red[tid][0] + igb[tid];
  } else if (tid < 2*NH_) {
    const int h = tid - NH_;
    fgp[((size_t)(b*NH_ + h))*S_ + s] = red[4+h][0] + fgb[h];
  }
}

// ---------------------------------------------------------------------------
// Kernel 2: F[s] = inclusive cumsum over s of log_sigmoid(fgate_preact).
// ---------------------------------------------------------------------------
__global__ void scan_kernel(const float* __restrict__ fgp, float* __restrict__ F)
{
  const int bh = blockIdx.x;             // B*NH = 8
  const int t  = threadIdx.x;            // 256
  const size_t base = (size_t)bh * S_;

  float loc[8];
  float run = 0.f;
  #pragma unroll
  for (int i = 0; i < 8; ++i) {
    const float x  = fgp[base + t*8 + i];
    const float ls = fminf(x, 0.f) - log1pf(expf(-fabsf(x)));   // stable log-sigmoid
    run += ls;
    loc[i] = run;
  }
  __shared__ float ssum[256];
  ssum[t] = run;
  __syncthreads();
  for (int off = 1; off < 256; off <<= 1) {
    const float vv = (t >= off) ? ssum[t-off] : 0.f;
    __syncthreads();
    ssum[t] += vv;
    __syncthreads();
  }
  const float excl = ssum[t] - run;
  #pragma unroll
  for (int i = 0; i < 8; ++i) F[base + t*8 + i] = excl + loc[i];
}

// ---------------------------------------------------------------------------
// Kernel 3: repack to TDM-friendly f16.
//   qh,kh : (B,NH,S,DH) f16 (q pre-scaled by 1/sqrt(DH))
//   vhT   : (B,NH,DH,S) f16 (pre-transposed so TDM tiles land as LDS [d][t])
// ---------------------------------------------------------------------------
__global__ void convert_kernel(const float* __restrict__ q, const float* __restrict__ k,
                               const float* __restrict__ v,
                               _Float16* __restrict__ qh, _Float16* __restrict__ kh,
                               _Float16* __restrict__ vhT)
{
  __shared__ __align__(16) _Float16 Vt[64][264];    // padded to dodge bank conflicts
  const int s0  = blockIdx.x * 64;
  const int bh  = blockIdx.y;
  const int b   = bh >> 2, h = bh & 3;
  const int tid = threadIdx.x;                      // 256

  #pragma unroll
  for (int i = 0; i < 16; ++i) {                    // 64 rows * 64 float4 / 256 thr
    const int lin = i*256 + tid;
    const int r = lin >> 6, c4 = lin & 63;
    const size_t src = ((size_t)(b*S_ + s0 + r))*H_ + (size_t)h*DH_ + c4*4;
    const size_t dst = ((size_t)bh*S_ + s0 + r)*DH_ + c4*4;
    const float4 qx = *(const float4*)(q + src);
    v4h qy;
    qy[0] = (_Float16)(qx.x*0.0625f); qy[1] = (_Float16)(qx.y*0.0625f);
    qy[2] = (_Float16)(qx.z*0.0625f); qy[3] = (_Float16)(qx.w*0.0625f);
    *(v4h*)&qh[dst] = qy;
    const float4 kx = *(const float4*)(k + src);
    v4h ky;
    ky[0] = (_Float16)kx.x; ky[1] = (_Float16)kx.y;
    ky[2] = (_Float16)kx.z; ky[3] = (_Float16)kx.w;
    *(v4h*)&kh[dst] = ky;
    const float4 vx = *(const float4*)(v + src);
    v4h vy;
    vy[0] = (_Float16)vx.x; vy[1] = (_Float16)vx.y;
    vy[2] = (_Float16)vx.z; vy[3] = (_Float16)vx.w;
    *(v4h*)&Vt[r][c4*4] = vy;
  }
  __syncthreads();
  // transposed write-out: 8 lanes cover one d-row's 64 s-values (128B contiguous)
  const int wave = tid >> 5, lane = tid & 31;
  const int dl = lane >> 3, sl = lane & 7;
  #pragma unroll
  for (int it = 0; it < 8; ++it) {
    const int d = it*32 + wave*4 + dl;
    v8h o;
    #pragma unroll
    for (int j = 0; j < 8; ++j) o[j] = Vt[sl*8 + j][d];
    *(v8h*)&vhT[((size_t)bh*DH_ + d)*S_ + s0 + sl*8] = o;
  }
}

// ---------------------------------------------------------------------------
// TDM: 2D tensor_load_to_lds via inline asm (portable across toolchains).
// D# group0 = {count=1, lds_addr, global_addr, type=2}; group1 per ISA 8.4.
// ---------------------------------------------------------------------------
__device__ __forceinline__ u32 lds_off(const void* p) { return (u32)(uintptr_t)p; }

__device__ __forceinline__ void tdm_load_2d(u32 lds_addr, const _Float16* gp,
                                            u32 td0, u32 td1, u32 tile0, u32 tile1,
                                            u32 stride0)
{
  const uint64_t ga = (uint64_t)(uintptr_t)gp;
  u32x4 g0;
  g0[0] = 1u;                                           // count=1 (valid descriptor)
  g0[1] = lds_addr;                                     // LDS byte address
  g0[2] = (u32)ga;                                      // global_addr[31:0]
  g0[3] = ((u32)(ga >> 32) & 0x01FFFFFFu) | (2u << 30); // global_addr[56:32] | type=2
  u32x8 g1;
  g1[0] = (1u << 16);                                   // data_size = 2 bytes
  g1[1] = (td0 & 0xFFFFu) << 16;                        // tensor_dim0[15:0]
  g1[2] = (td0 >> 16) | ((td1 & 0xFFFFu) << 16);        // tensor_dim0[31:16] | tensor_dim1[15:0]
  g1[3] = (td1 >> 16) | (tile0 << 16);                  // tensor_dim1[31:16] | tile_dim0
  g1[4] = tile1;                                        // tile_dim1 (tile_dim2 = 0)
  g1[5] = stride0;                                      // tensor_dim0_stride[31:0]
  g1[6] = 0u;                                           // stride0 hi / dim1_stride lo
  g1[7] = 0u;
  asm volatile("tensor_load_to_lds %0, %1" :: "s"(g0), "s"(g1) : "memory");
}

// ---------------------------------------------------------------------------
// Kernel 4: flash-style stabilized mLSTM attention + fused multihead LayerNorm.
// ---------------------------------------------------------------------------
__device__ __forceinline__ v16h frag_ab(const _Float16* p0, const _Float16* p1)
{
  const v8h lo = *(const v8h*)p0;
  const v8h hi = *(const v8h*)p1;
  v16h r;
  #pragma unroll
  for (int i = 0; i < 8; ++i) { r[i] = lo[i]; r[i+8] = hi[i]; }
  return r;
}
__device__ __forceinline__ float redmax16(float x)
{
  #pragma unroll
  for (int m = 1; m < 16; m <<= 1) x = fmaxf(x, __shfl_xor(x, m, 16));
  return x;
}
__device__ __forceinline__ float redsum16(float x)
{
  #pragma unroll
  for (int m = 1; m < 16; m <<= 1) x += __shfl_xor(x, m, 16);
  return x;
}

__launch_bounds__(128, 1)
__global__ void mlstm_flash_kernel(const _Float16* __restrict__ qh,
                                   const _Float16* __restrict__ kh,
                                   const _Float16* __restrict__ vhT,
                                   const float* __restrict__ F,
                                   const float* __restrict__ igp,
                                   const float* __restrict__ oscale,
                                   float* __restrict__ out)
{
  __shared__ __align__(16) _Float16 Qs[64*DH_];       // 32 KB
  __shared__ __align__(16) _Float16 Ks[2][32*DH_];    // 2 x 16 KB, K tile row-major
  __shared__ __align__(16) _Float16 Vs[2][DH_*32];    // 2 x 16 KB, V tile [d][t]
  __shared__ __align__(16) _Float16 Ps[4*16*32];      // 4 KB, per-wave P stash

  const int rb   = blockIdx.x;           // 0..31 row blocks of 64
  const int bh   = blockIdx.y;           // 0..7
  const int b    = bh >> 2, h = bh & 3;
  const int row0 = rb * 64;
  const int tid  = threadIdx.x;
  const int wave = tid >> 5;
  const int lane = tid & 31;
  const int n    = lane & 15;
  const int half = lane >> 4;
  const int halfOff = half * 8;
  const int wr0  = row0 + wave*16;

  const size_t bhS   = (size_t)bh * S_;
  const size_t headE = (size_t)bh * S_ * DH_;   // head base in f16 elements
  const int nTb = row0/32 + 2;                  // causal: key tiles 0 .. row0+63

  // ---- prologue: TDM Q tile + first K/V tiles, wave0 drives the DMA ----
  if (wave == 0) {
    tdm_load_2d(lds_off(Qs),    qh  + headE + (size_t)row0*DH_, DH_, S_, DH_, 64, DH_);
    tdm_load_2d(lds_off(Ks[0]), kh  + headE,                    DH_, S_, DH_, 32, DH_);
    tdm_load_2d(lds_off(Vs[0]), vhT + headE,                    S_, DH_, 32, DH_, S_);
    __builtin_amdgcn_s_wait_tensorcnt(0);
  }
  __syncthreads();

  // per-row gating state (row = wr0 + g + halfOff)
  float Frow[8], mrow[8], rsum[8];
  #pragma unroll
  for (int g = 0; g < 8; ++g) {
    Frow[g] = F[bhS + wr0 + g + halfOff];
    mrow[g] = -1e30f;
    rsum[g] = 0.f;
  }
  v8f acc[16];
  {
    const v8f z = {};
    #pragma unroll
    for (int dt = 0; dt < 16; ++dt) acc[dt] = z;
  }

  for (int tb = 0; tb < nTb; ++tb) {
    const int t0  = tb*32;
    const int buf = tb & 1;

    // kick DMA for next tile (overlaps with this tile's WMMA work)
    if (wave == 0 && tb + 1 < nTb) {
      const int t1 = t0 + 32;
      tdm_load_2d(lds_off(Ks[buf^1]), kh  + headE + (size_t)t1*DH_, DH_, S_, DH_, 32, DH_);
      tdm_load_2d(lds_off(Vs[buf^1]), vhT + headE + t1,             S_, DH_, 32, DH_, S_);
    }

    if (t0 <= wr0 + 15) {                // this wave has unmasked work in this tile
      // ---- GEMM1: S = (Q/sqrt(dh)) K^T, 16x32 tile as two 16x16 WMMAs ----
      v8f s0 = {}, s1 = {};
      #pragma unroll
      for (int c = 0; c < 8; ++c) {      // K = 256 in chunks of 32
        const _Float16* qb = &Qs[(wave*16 + n)*DH_ + c*32];
        const v16h a  = frag_ab(qb + halfOff, qb + 16 + halfOff);
        const _Float16* k0 = &Ks[buf][n*DH_ + c*32];
        const v16h b0 = frag_ab(k0 + halfOff, k0 + 16 + halfOff);
        const _Float16* k1 = &Ks[buf][(16+n)*DH_ + c*32];
        const v16h b1 = frag_ab(k1 + halfOff, k1 + 16 + halfOff);
        s0 = __builtin_amdgcn_wmma_f32_16x16x32_f16(false, a, false, b0, (short)0, s0, false, false);
        s1 = __builtin_amdgcn_wmma_f32_16x16x32_f16(false, a, false, b1, (short)0, s1, false, false);
      }

      // ---- gating + online stabilization ----
      const float Fc0 = F  [bhS + t0 + n];
      const float Ic0 = igp[bhS + t0 + n];
      const float Fc1 = F  [bhS + t0 + 16 + n];
      const float Ic1 = igp[bhS + t0 + 16 + n];
      const int col0 = t0 + n, col1 = t0 + 16 + n;

      float scl[8], mnew[8], l0[8], l1[8];
      bool  vv0[8], vv1[8];
      bool  noRescale = true;
      #pragma unroll
      for (int g = 0; g < 8; ++g) {
        const int row = wr0 + g + halfOff;
        vv0[g] = (row >= col0);
        vv1[g] = (row >= col1);
        l0[g] = Frow[g] - Fc0 + Ic0;     // log_D = F[s] - F[t] + ig[t]
        l1[g] = Frow[g] - Fc1 + Ic1;
        float tm = fmaxf(vv0[g] ? l0[g] : -1e30f, vv1[g] ? l1[g] : -1e30f);
        tm = redmax16(tm);
        mnew[g] = fmaxf(mrow[g], tm);
        scl[g]  = __expf(mrow[g] - mnew[g]);
        noRescale = noRescale && (scl[g] == 1.f);
        mrow[g] = mnew[g];
      }

      // P = S * exp(logD - m); rowsum update; stash P (C/D -> A layout via LDS)
      const int pbase = wave*16*32;
      #pragma unroll
      for (int g = 0; g < 8; ++g) {
        const float p0 = vv0[g] ? s0[g]*__expf(l0[g]-mnew[g]) : 0.f;
        const float p1 = vv1[g] ? s1[g]*__expf(l1[g]-mnew[g]) : 0.f;
        rsum[g] = rsum[g]*scl[g] + redsum16(p0 + p1);
        const int r = g + halfOff;
        Ps[pbase + r*32 + n]      = (_Float16)p0;
        Ps[pbase + r*32 + 16 + n] = (_Float16)p1;
      }
      if (!__all(noRescale)) {           // common steady state skips 128 multiplies
        #pragma unroll
        for (int dt = 0; dt < 16; ++dt) {
          #pragma unroll
          for (int g = 0; g < 8; ++g) acc[dt][g] *= scl[g];
        }
      }

      // ---- GEMM2: acc += P @ V (K=32, 16 output d-tiles) ----
      const _Float16* pa = &Ps[pbase + n*32];
      const v16h ap = frag_ab(pa + halfOff, pa + 16 + halfOff);
      #pragma unroll
      for (int dt = 0; dt < 16; ++dt) {
        const _Float16* vb = &Vs[buf][(dt*16 + n)*32];
        const v16h bv = frag_ab(vb + halfOff, vb + 16 + halfOff);
        acc[dt] = __builtin_amdgcn_wmma_f32_16x16x32_f16(false, ap, false, bv, (short)0, acc[dt], false, false);
      }
    }

    if (wave == 0) __builtin_amdgcn_s_wait_tensorcnt(0);  // next tile landed
    __syncthreads();
  }

  // ---- epilogue: normalizer, fused per-head LayerNorm over DH=256, store ----
  const size_t headoff = (size_t)h * DH_;
  float norm[8];
  #pragma unroll
  for (int g = 0; g < 8; ++g)
    norm[g] = fmaxf(fabsf(rsum[g]), __expf(-mrow[g])) + 1e-6f;

  float sum[8] = {0,0,0,0,0,0,0,0}, sq[8] = {0,0,0,0,0,0,0,0};
  #pragma unroll
  for (int dt = 0; dt < 16; ++dt) {
    #pragma unroll
    for (int g = 0; g < 8; ++g) {
      const float x = acc[dt][g] / norm[g];
      acc[dt][g] = x;
      sum[g] += x;
      sq[g]  += x*x;
    }
  }
  float mean[8], rstd[8];
  #pragma unroll
  for (int g = 0; g < 8; ++g) {
    const float s  = redsum16(sum[g]);
    const float s2 = redsum16(sq[g]);
    const float mu = s * (1.f/256.f);
    const float var = s2 * (1.f/256.f) - mu*mu;
    mean[g] = mu;
    rstd[g] = rsqrtf(var + 1e-6f);
  }
  #pragma unroll
  for (int dt = 0; dt < 16; ++dt) {
    const float sc = oscale[headoff + dt*16 + n];
    #pragma unroll
    for (int g = 0; g < 8; ++g) {
      const int srow = wr0 + g + halfOff;
      out[((size_t)(b*S_ + srow))*H_ + headoff + dt*16 + n] =
          (acc[dt][g] - mean[g]) * rstd[g] * sc;
    }
  }
}

// ---------------------------------------------------------------------------
extern "C" void kernel_launch(void* const* d_in, const int* in_sizes, int n_in,
                              void* d_out, int out_size, void* d_ws, size_t ws_size,
                              hipStream_t stream)
{
  const float* q   = (const float*)d_in[0];
  const float* k   = (const float*)d_in[1];
  const float* v   = (const float*)d_in[2];
  const float* igk = (const float*)d_in[3];
  const float* igb = (const float*)d_in[4];
  const float* fgk = (const float*)d_in[5];
  const float* fgb = (const float*)d_in[6];
  const float* osc = (const float*)d_in[7];
  float* out = (float*)d_out;

  const size_t nBHS = (size_t)B_*NH_*S_;        // 16384
  const size_t nE   = (size_t)B_*NH_*S_*DH_;    // 4M f16 elements per tensor
  float*    igp = (float*)d_ws;                 // (B,NH,S) igate preacts
  float*    fgp = igp + nBHS;                   // (B,NH,S) fgate preacts
  float*    F   = fgp + nBHS;                   // (B,NH,S) cumsum log-sigmoid
  _Float16* qh  = (_Float16*)(F + nBHS);        // (B,NH,S,DH)  f16, pre-scaled
  _Float16* kh  = qh + nE;                      // (B,NH,S,DH)  f16
  _Float16* vhT = kh + nE;                      // (B,NH,DH,S)  f16, transposed

  gates_kernel  <<<dim3(B_*S_),        dim3(128), 0, stream>>>(q, k, v, igk, igb, fgk, fgb, igp, fgp);
  scan_kernel   <<<dim3(B_*NH_),       dim3(256), 0, stream>>>(fgp, F);
  convert_kernel<<<dim3(S_/64, B_*NH_), dim3(256), 0, stream>>>(q, k, v, qh, kh, vhT);
  mlstm_flash_kernel<<<dim3(S_/64, B_*NH_), dim3(128), 0, stream>>>(qh, kh, vhT, F, igp, osc, out);
}
